// FuseMoEFusionLayer_39745627357230
// MI455X (gfx1250) — matrix operand order repaired and checked by reading, not compiled, not run on hardware
//
#include <hip/hip_runtime.h>
#include <hip/hip_bf16.h>
#include <math.h>

#define NTOK   8192
#define DMODEL 1024
#define NEXP   8
#define HDIM   1024
#define RHDIM  256
#define INDIM  (9*DMODEL+3)   // 9219
#define KP     9248           // 289*32, zero-padded K for x / w1t / r1t
#define LN_EPS 1e-5f

typedef __attribute__((ext_vector_type(16))) __bf16 bf16x16;
typedef __attribute__((ext_vector_type(8)))  float  f32x8;

__device__ __forceinline__ unsigned short f2bf(float f) {
  unsigned u = __float_as_uint(f);
  unsigned r = (u >> 16) & 1u;
  u += 0x7fffu + r;                      // round-to-nearest-even
  return (unsigned short)(u >> 16);
}
__device__ __forceinline__ float bf2f(unsigned short h) {
  return __uint_as_float(((unsigned)h) << 16);
}
__device__ __forceinline__ float gelu_erf(float x) {
  return 0.5f * x * (1.0f + erff(x * 0.70710678118654752f));
}

// CDNA5 async copy: one 128-bit chunk global -> LDS per lane (ASYNCcnt)
__device__ __forceinline__ void async_cp16(const void* gsrc, void* ldst) {
  unsigned loff = (unsigned)(size_t)ldst;   // low 32 bits of flat LDS addr = LDS offset
  asm volatile("global_load_async_to_lds_b128 %0, %1, off"
               :: "v"(loff), "v"(gsrc) : "memory");
}
__device__ __forceinline__ void wait_async0() {
  asm volatile("s_wait_asynccnt 0x0" ::: "memory");
}

// block-wide sum of two values; 256 threads = 8 wave32 waves
__device__ __forceinline__ void block_reduce2(float& a, float& b, float* sbuf) {
  #pragma unroll
  for (int o = 16; o > 0; o >>= 1) {
    a += __shfl_down(a, o, 32);
    b += __shfl_down(b, o, 32);
  }
  int lane = threadIdx.x & 31, w = threadIdx.x >> 5;
  if (lane == 0) { sbuf[w] = a; sbuf[8 + w] = b; }
  __syncthreads();
  if (threadIdx.x == 0) {
    float sa = 0.f, sb = 0.f;
    #pragma unroll
    for (int i = 0; i < 8; ++i) { sa += sbuf[i]; sb += sbuf[8 + i]; }
    sbuf[0] = sa; sbuf[8] = sb;
  }
  __syncthreads();
  a = sbuf[0]; b = sbuf[8];
  __syncthreads();
}

// ---------------- kernel 1: build fusion input x (bf16) + residual ----------
__global__ __launch_bounds__(256) void build_x_kernel(
    const float* __restrict__ z1, const float* __restrict__ z2,
    const float* __restrict__ z3, const float* __restrict__ mask,
    const float* __restrict__ ln_g, const float* __restrict__ ln_b,
    const float* __restrict__ lnp_g, const float* __restrict__ lnp_b,
    unsigned short* __restrict__ xb, float* __restrict__ zres)
{
  __shared__ float s1[DMODEL], s2[DMODEL], s3[DMODEL];
  __shared__ float red[16];
  const int n = blockIdx.x, tid = threadIdx.x;
  unsigned short* xrow = xb + (size_t)n * KP;

  auto do_ln = [&](const float* __restrict__ zp, float* __restrict__ sdst) {
    float v[4]; float sum = 0.f, sq = 0.f;
    #pragma unroll
    for (int i = 0; i < 4; ++i) {
      v[i] = zp[(size_t)n * DMODEL + tid + i * 256];
      sum += v[i]; sq += v[i] * v[i];
    }
    block_reduce2(sum, sq, red);
    float mu = sum * (1.f / DMODEL);
    float var = sq * (1.f / DMODEL) - mu * mu;
    float rs = rsqrtf(var + LN_EPS);
    #pragma unroll
    for (int i = 0; i < 4; ++i) {
      int j = tid + i * 256;
      sdst[j] = (v[i] - mu) * rs * ln_g[j] + ln_b[j];
    }
  };
  do_ln(z1, s1); do_ln(z2, s2); do_ln(z3, s3);
  __syncthreads();

  float m0 = mask[n*3], m1 = mask[n*3+1], m2 = mask[n*3+2];
  float denom = fmaxf(m0 + m1 + m2, 1.0f);
  #pragma unroll
  for (int i = 0; i < 4; ++i) {
    int j = tid + i * 256;
    zres[(size_t)n * DMODEL + j] = (m0*s1[j] + m1*s2[j] + m2*s3[j]) / denom;
  }
  #pragma unroll
  for (int i = 0; i < 4; ++i) {
    int j = tid + i * 256;
    xrow[0*DMODEL + j] = f2bf(s1[j]);
    xrow[1*DMODEL + j] = f2bf(s2[j]);
    xrow[2*DMODEL + j] = f2bf(s3[j]);
  }
  auto do_pair = [&](const float* __restrict__ a, const float* __restrict__ b,
                     bool prod, int seg) {
    float t[4]; float sum = 0.f, sq = 0.f;
    #pragma unroll
    for (int i = 0; i < 4; ++i) {
      int j = tid + i * 256;
      float x = prod ? (a[j] * b[j]) : (a[j] - b[j]);
      t[i] = x; sum += x; sq += x * x;
    }
    block_reduce2(sum, sq, red);
    float mu = sum * (1.f / DMODEL);
    float var = sq * (1.f / DMODEL) - mu * mu;
    float rs = rsqrtf(var + LN_EPS);
    #pragma unroll
    for (int i = 0; i < 4; ++i) {
      int j = tid + i * 256;
      xrow[seg*DMODEL + j] = f2bf((t[i] - mu) * rs * lnp_g[j] + lnp_b[j]);
    }
  };
  do_pair(s1, s2, false, 3);
  do_pair(s1, s3, false, 4);
  do_pair(s2, s3, false, 5);
  do_pair(s1, s2, true,  6);
  do_pair(s1, s3, true,  7);
  do_pair(s2, s3, true,  8);

  if (tid < 32) {
    int c = 9 * DMODEL + tid;
    float v = (tid < 3) ? mask[n*3 + tid] : 0.f;
    xrow[c] = f2bf(v);
  }
}

// ------- weight prep: transpose [R][C] f32 -> [C][RP] bf16 (zero-padded) ----
__global__ __launch_bounds__(256) void transpose_bf16_kernel(
    const float* __restrict__ in, unsigned short* __restrict__ out,
    int R, int C, int RP, size_t inMatStride, size_t outMatStride)
{
  __shared__ float tile[32][33];
  const int tid = threadIdx.x;
  const float* src = in + (size_t)blockIdx.z * inMatStride;
  unsigned short* dst = out + (size_t)blockIdx.z * outMatStride;
  const int c0 = blockIdx.x * 32, r0 = blockIdx.y * 32;
  #pragma unroll
  for (int p = 0; p < 4; ++p) {
    int idx = p * 256 + tid;
    int rr = idx >> 5, cc = idx & 31;
    float v = 0.f;
    if (r0 + rr < R) v = src[(size_t)(r0 + rr) * C + c0 + cc];
    tile[rr][cc] = v;
  }
  __syncthreads();
  #pragma unroll
  for (int p = 0; p < 4; ++p) {
    int idx = p * 256 + tid;
    int cc = idx >> 5, rr = idx & 31;
    dst[(size_t)(c0 + cc) * RP + r0 + rr] = f2bf(tile[rr][cc]);
  }
}

// ---------------- WMMA GEMM: C = epilogue(A_bf16 @ Bt_bf16^T) ---------------
// Block tile 128(M) x 256(N), BK=32; 8 waves, each 64x64 (4x4 WMMA frags).
// A: [M][ldA] bf16 row-major (k contiguous); Bt: [Ncols][ldBt] bf16 (k contig)
// MODE 0: router GEMM1 -> gelu(+r_b1), bf16 out
// MODE 1: expert GEMM1 -> gelu(+e_b1) * gate[n,e], bf16 out (e = col>>10)
// MODE 2: expert GEMM2 -> + sum_e gate*e_b2 + zres, f32 out
template<int MODE>
__global__ __launch_bounds__(256) void gemm_wmma_kernel(
    const unsigned short* __restrict__ A, int ldA,
    const unsigned short* __restrict__ Bt, int ldBt, int Ksteps,
    unsigned short* __restrict__ Cbf, float* __restrict__ Cf,
    const float* __restrict__ bias, const float* __restrict__ gate,
    const float* __restrict__ e_b2, const float* __restrict__ zres, int ldC)
{
  __shared__ unsigned short As[2][128*32];   // [m][k] 16KB
  __shared__ unsigned short Bs[2][256*32];   // [n][k] 32KB
  const int tid = threadIdx.x;
  const int m0 = blockIdx.x * 128;
  const int n0 = blockIdx.y * 256;

  // async-copy 128x32 A tile (2 chunks/thread) + 256x32 B tile (4 chunks/thread)
  auto issue_tiles = [&](int buf, int s) {
    int k0 = s * 32;
    #pragma unroll
    for (int p = 0; p < 2; ++p) {
      int idx = p * 256 + tid;
      int row = idx >> 2, ch = (idx & 3) * 8;
      async_cp16(A + (size_t)(m0 + row) * ldA + k0 + ch, &As[buf][row * 32 + ch]);
    }
    #pragma unroll
    for (int p = 0; p < 4; ++p) {
      int idx = p * 256 + tid;
      int row = idx >> 2, ch = (idx & 3) * 8;
      async_cp16(Bt + (size_t)(n0 + row) * ldBt + k0 + ch, &Bs[buf][row * 32 + ch]);
    }
  };

  const int lane = tid & 31;
  const int wave = tid >> 5;
  const int wm0 = (wave & 1) * 64;     // 2x4 wave grid -> 64x64 per wave
  const int wn0 = (wave >> 1) * 64;
  const int lhi = lane >> 4;
  const int l15 = lane & 15;

  f32x8 acc[4][4];
  const f32x8 zero8 = {0.f,0.f,0.f,0.f,0.f,0.f,0.f,0.f};
  #pragma unroll
  for (int mi = 0; mi < 4; ++mi)
    #pragma unroll
    for (int ni = 0; ni < 4; ++ni) acc[mi][ni] = zero8;

  issue_tiles(0, 0);
  int buf = 0;
  for (int s = 0; s < Ksteps; ++s) {
    wait_async0();          // this wave's tile data landed in LDS
    __syncthreads();        // all waves' tile data landed
    if (s + 1 < Ksteps) issue_tiles(buf ^ 1, s + 1);
    union Frag { bf16x16 v; uint4 q[2]; } a[4], b[4];
    // A frag: lanes 0-15 hold K 0..7 & 16..23; lanes 16-31 hold K 8..15 & 24..31
    #pragma unroll
    for (int mi = 0; mi < 4; ++mi) {
      const unsigned short* p = &As[buf][(wm0 + mi*16 + l15) * 32 + (lhi ? 8 : 0)];
      a[mi].q[0] = *reinterpret_cast<const uint4*>(p);
      a[mi].q[1] = *reinterpret_cast<const uint4*>(p + 16);
    }
    // B frag: lanes 0-15 hold K 0..7 & 16..23; lanes 16-31 hold K 8..15 & 24..31 of column n
    #pragma unroll
    for (int ni = 0; ni < 4; ++ni) {
      const unsigned short* p = &Bs[buf][(wn0 + ni*16 + l15) * 32 + (lhi ? 8 : 0)];
      b[ni].q[0] = *reinterpret_cast<const uint4*>(p);
      b[ni].q[1] = *reinterpret_cast<const uint4*>(p + 16);
    }
    #pragma unroll
    for (int mi = 0; mi < 4; ++mi)
      #pragma unroll
      for (int ni = 0; ni < 4; ++ni)
        acc[mi][ni] = __builtin_amdgcn_wmma_f32_16x16x32_bf16(
            false, a[mi].v, false, b[ni].v, (short)0, acc[mi][ni], false, false);
    __syncthreads();        // everyone done reading buf before next async writes
    buf ^= 1;
  }

  #pragma unroll
  for (int mi = 0; mi < 4; ++mi)
    #pragma unroll
    for (int ni = 0; ni < 4; ++ni)
      #pragma unroll
      for (int r = 0; r < 8; ++r) {
        int row = m0 + wm0 + mi*16 + r + (lhi << 3);
        int col = n0 + wn0 + ni*16 + l15;
        float v = acc[mi][ni][r];
        if (MODE == 0) {
          v = gelu_erf(v + bias[col]);
          Cbf[(size_t)row * ldC + col] = f2bf(v);
        } else if (MODE == 1) {
          v = gelu_erf(v + bias[col]);          // e_b1 flat index == col
          int e = col >> 10;
          v *= gate[row * NEXP + e];
          Cbf[(size_t)row * ldC + col] = f2bf(v);
        } else {
          float g2 = 0.f;
          #pragma unroll
          for (int e = 0; e < NEXP; ++e)
            g2 += gate[row * NEXP + e] * e_b2[e * DMODEL + col];
          v += g2 + zres[(size_t)row * DMODEL + col];
          Cf[(size_t)row * ldC + col] = v;
        }
      }
}

// ---------------- router second matmul + softmax -> gate ----------
__global__ __launch_bounds__(256) void router2_kernel(
    const unsigned short* __restrict__ hr, const float* __restrict__ r_w2,
    const float* __restrict__ r_b2, const float* __restrict__ log_temp,
    float* __restrict__ gate)
{
  int n = blockIdx.x * 256 + threadIdx.x;
  float acc[NEXP];
  #pragma unroll
  for (int e = 0; e < NEXP; ++e) acc[e] = r_b2[e];
  const unsigned short* h = hr + (size_t)n * RHDIM;
  for (int k = 0; k < RHDIM; ++k) {
    float hv = bf2f(h[k]);
    #pragma unroll
    for (int e = 0; e < NEXP; ++e) acc[e] += hv * r_w2[k * NEXP + e];
  }
  float temp = fminf(fmaxf(expf(log_temp[0]), 0.001f), 100.f);
  float it = 1.0f / temp;
  float mx = -3.4e38f;
  #pragma unroll
  for (int e = 0; e < NEXP; ++e) { acc[e] *= it; mx = fmaxf(mx, acc[e]); }
  float sum = 0.f;
  #pragma unroll
  for (int e = 0; e < NEXP; ++e) { acc[e] = expf(acc[e] - mx); sum += acc[e]; }
  float inv = 1.0f / sum;
  #pragma unroll
  for (int e = 0; e < NEXP; ++e) gate[(size_t)n * NEXP + e] = acc[e] * inv;
}

// ---------------- output LayerNorm ----------
__global__ __launch_bounds__(256) void out_ln_kernel(
    const float* __restrict__ z, const float* __restrict__ g,
    const float* __restrict__ b, float* __restrict__ out)
{
  __shared__ float red[16];
  const int n = blockIdx.x, tid = threadIdx.x;
  float v[4]; float sum = 0.f, sq = 0.f;
  #pragma unroll
  for (int i = 0; i < 4; ++i) {
    v[i] = z[(size_t)n * DMODEL + tid + i * 256];
    sum += v[i]; sq += v[i] * v[i];
  }
  block_reduce2(sum, sq, red);
  float mu = sum * (1.f / DMODEL);
  float var = sq * (1.f / DMODEL) - mu * mu;
  float rs = rsqrtf(var + LN_EPS);
  #pragma unroll
  for (int i = 0; i < 4; ++i) {
    int j = tid + i * 256;
    out[(size_t)n * DMODEL + j] = (v[i] - mu) * rs * g[j] + b[j];
  }
}

extern "C" void kernel_launch(void* const* d_in, const int* in_sizes, int n_in,
                              void* d_out, int out_size, void* d_ws, size_t ws_size,
                              hipStream_t stream) {
  const float* z1    = (const float*)d_in[0];
  const float* z2    = (const float*)d_in[1];
  const float* z3    = (const float*)d_in[2];
  const float* mask  = (const float*)d_in[3];
  const float* ln_g  = (const float*)d_in[4];
  const float* ln_b  = (const float*)d_in[5];
  const float* lnp_g = (const float*)d_in[6];
  const float* lnp_b = (const float*)d_in[7];
  const float* r_w1  = (const float*)d_in[8];
  const float* r_b1  = (const float*)d_in[9];
  const float* r_w2  = (const float*)d_in[10];
  const float* r_b2  = (const float*)d_in[11];
  const float* log_t = (const float*)d_in[12];
  const float* e_w1  = (const float*)d_in[13];
  const float* e_b1  = (const float*)d_in[14];
  const float* e_w2  = (const float*)d_in[15];
  const float* e_b2  = (const float*)d_in[16];
  const float* out_g = (const float*)d_in[17];
  const float* out_b = (const float*)d_in[18];
  (void)in_sizes; (void)n_in; (void)out_size; (void)ws_size;

  // workspace layout (~497 MB)
  char* ws = (char*)d_ws;
  auto al = [](size_t x) { return (x + 255) & ~(size_t)255; };
  size_t off = 0;
  unsigned short* xb   = (unsigned short*)(ws + off); off += al((size_t)NTOK * KP * 2);
  unsigned short* r1t  = (unsigned short*)(ws + off); off += al((size_t)RHDIM * KP * 2);
  unsigned short* w1t  = (unsigned short*)(ws + off); off += al((size_t)NEXP * HDIM * KP * 2);
  unsigned short* w2t  = (unsigned short*)(ws + off); off += al((size_t)DMODEL * NEXP * HDIM * 2);
  unsigned short* hr   = (unsigned short*)(ws + off); off += al((size_t)NTOK * RHDIM * 2);
  float*          gate = (float*)         (ws + off); off += al((size_t)NTOK * NEXP * 4);
  unsigned short* hb   = (unsigned short*)(ws + off); off += al((size_t)NTOK * NEXP * HDIM * 2);
  float*          zres = (float*)         (ws + off); off += al((size_t)NTOK * DMODEL * 4);
  float*          zacc = (float*)xb;  // xb is dead after expert GEMM1

  // 1. fused LayerNorm stack -> x (bf16, zero-padded K) + residual
  build_x_kernel<<<NTOK, 256, 0, stream>>>(z1, z2, z3, mask, ln_g, ln_b,
                                           lnp_g, lnp_b, xb, zres);
  // 2. weight prep (one-time per launch, ~20us of bandwidth)
  transpose_bf16_kernel<<<dim3(RHDIM/32, (INDIM+31)/32, 1), 256, 0, stream>>>(
      r_w1, r1t, INDIM, RHDIM, KP, 0, 0);
  transpose_bf16_kernel<<<dim3(HDIM/32, (INDIM+31)/32, NEXP), 256, 0, stream>>>(
      e_w1, w1t, INDIM, HDIM, KP, (size_t)INDIM*HDIM, (size_t)HDIM*KP);
  transpose_bf16_kernel<<<dim3(DMODEL/32, (NEXP*HDIM)/32, 1), 256, 0, stream>>>(
      e_w2, w2t, NEXP*HDIM, DMODEL, NEXP*HDIM, 0, 0);

  // 3. router GEMM1: hr = gelu(x @ r_w1 + r_b1)
  gemm_wmma_kernel<0><<<dim3(NTOK/128, RHDIM/256), 256, 0, stream>>>(
      xb, KP, r1t, KP, KP/32,
      hr, nullptr, r_b1, nullptr, nullptr, nullptr, RHDIM);
  // 4. router GEMM2 + softmax -> gate
  router2_kernel<<<NTOK/256, 256, 0, stream>>>(hr, r_w2, r_b2, log_t, gate);
  // 5. expert GEMM1 (all 8 experts as one 8192-wide N): h = gelu(x@W1+b1)*gate
  gemm_wmma_kernel<1><<<dim3(NTOK/128, (NEXP*HDIM)/256), 256, 0, stream>>>(
      xb, KP, w1t, KP, KP/32,
      hb, nullptr, e_b1, gate, nullptr, nullptr, NEXP*HDIM);
  // 6. expert GEMM2 collapsed: z = h' @ W2cat + sum_e gate*b2 + residual
  gemm_wmma_kernel<2><<<dim3(NTOK/128, DMODEL/256), 256, 0, stream>>>(
      hb, NEXP*HDIM, w2t, NEXP*HDIM, (NEXP*HDIM)/32,
      nullptr, zacc, nullptr, gate, e_b2, zres, DMODEL);
  // 7. output LayerNorm -> d_out
  out_ln_kernel<<<NTOK, 256, 0, stream>>>(zacc, out_g, out_b, (float*)d_out);
}
